// GAT_67551245631641
// MI455X (gfx1250) — compile-verified, matrix operand order checked by reading, not run on hardware
//
#include <hip/hip_runtime.h>
#include <hip/hip_bf16.h>

// ---------------------------------------------------------------------------
// 2-layer GAT for MI455X (gfx1250).  GEMM1 (100000x128 @ 128x128) runs on
// v_wmma_f32_16x16x32_f16; the graph scatter/gather phases use
// global_atomic_{max,min,add} which L2 services (entire working set < 192MB L2).
// ---------------------------------------------------------------------------

typedef _Float16 half8  __attribute__((ext_vector_type(8)));
typedef _Float16 v16h   __attribute__((ext_vector_type(16)));
typedef float    v8f    __attribute__((ext_vector_type(8)));

#define F128 128   // F_IN == HEADS*HID == 128
#define NHEAD 8
#define HIDD 16
#define NEG_SLOPE 0.2f

// -------------------- packing --------------------
__global__ void pack_x_f16(const float* __restrict__ x, _Float16* __restrict__ xh, int n) {
    int i = blockIdx.x * blockDim.x + threadIdx.x;
    if (i < n) xh[i] = (_Float16)x[i];
}

// W1 is [128][128] (Fin, out) row-major; store transposed as w1t[out][k]
__global__ void pack_w1t_f16(const float* __restrict__ W1, _Float16* __restrict__ w1t) {
    int i = blockIdx.x * blockDim.x + threadIdx.x;     // i = o*128 + k
    if (i < F128 * F128) {
        int o = i >> 7, k = i & 127;
        w1t[i] = (_Float16)W1[k * F128 + o];
    }
}

// -------------------- GEMM1 via WMMA --------------------
// grid.x = N/16 node tiles; block = 256 (8 waves); wave w covers output cols [16w,16w+16)
__global__ void gemm1_wmma(const _Float16* __restrict__ xh,
                           const _Float16* __restrict__ w1t,
                           const float* __restrict__ b1,
                           float* __restrict__ h1) {
    int wave  = threadIdx.x >> 5;
    int lane  = threadIdx.x & 31;
    int half_ = lane >> 4;
    int lm    = lane & 15;
    int mbase = blockIdx.x * 16;

    const _Float16* pa = xh  + (size_t)(mbase + lm) * F128;        // row of A
    const _Float16* pb = w1t + (size_t)(wave * 16 + lm) * F128;    // col of B (W1^T row)

    v8f c = {};
#pragma unroll
    for (int k0 = 0; k0 < F128; k0 += 32) {
        half8 alo = *(const half8*)(pa + k0 +      half_ * 8);
        half8 ahi = *(const half8*)(pa + k0 + 16 + half_ * 8);
        half8 blo = *(const half8*)(pb + k0 +      half_ * 8);
        half8 bhi = *(const half8*)(pb + k0 + 16 + half_ * 8);
        v16h a, b;
#pragma unroll
        for (int i = 0; i < 8; ++i) {
            a[i] = alo[i]; a[i + 8] = ahi[i];
            b[i] = blo[i]; b[i + 8] = bhi[i];
        }
        c = __builtin_amdgcn_wmma_f32_16x16x32_f16(false, a, false, b,
                                                   (short)0, c, false, false);
    }
    int   ncol = wave * 16 + lm;
    float bias = b1[ncol];
#pragma unroll
    for (int r = 0; r < 8; ++r) {
        int m = mbase + r + 8 * half_;
        h1[(size_t)m * F128 + ncol] = c[r] + bias;
    }
}

// -------------------- per-node attention coefficients (layer 1) --------------------
__global__ void att1_coeff(const float* __restrict__ h1,
                           const float* __restrict__ as_w, const float* __restrict__ ad_w,
                           float* __restrict__ a_src, float* __restrict__ a_dst, int Nn) {
    int i = blockIdx.x * blockDim.x + threadIdx.x;   // n*8 + head
    if (i >= Nn * NHEAD) return;
    int n = i >> 3, hh = i & 7;
    const float* ph = h1 + (size_t)n * F128 + hh * HIDD;
    float s = 0.f, d = 0.f;
#pragma unroll
    for (int k = 0; k < HIDD; ++k) {
        float v = ph[k];
        s += v * as_w[hh * HIDD + k];
        d += v * ad_w[hh * HIDD + k];
    }
    a_src[i] = s; a_dst[i] = d;
}

// -------------------- edge helpers --------------------
__device__ __forceinline__ void edge_sd(const int* __restrict__ ei, int E, int e,
                                        int& s, int& d) {
    if (e < E) { s = ei[e]; d = ei[E + e]; }
    else       { s = e - E; d = s; }          // self-loops appended
}
__device__ __forceinline__ float leaky(float v) { return v >= 0.f ? v : NEG_SLOPE * v; }
__device__ __forceinline__ void atomicMaxF(float* addr, float v) {
    if (v >= 0.f) atomicMax((int*)addr, __float_as_int(v));
    else          atomicMin((unsigned int*)addr, __float_as_uint(v));
}

// -------------------- layer-1 edge passes --------------------
__global__ void edge_max1(const int* __restrict__ ei, int E, int EP,
                          const float* __restrict__ a_src, const float* __restrict__ a_dst,
                          float* __restrict__ emax) {
    int i = blockIdx.x * blockDim.x + threadIdx.x;
    if (i >= EP * NHEAD) return;
    int e = i >> 3, hh = i & 7, s, d;
    edge_sd(ei, E, e, s, d);
    atomicMaxF(&emax[d * NHEAD + hh], leaky(a_src[s * NHEAD + hh] + a_dst[d * NHEAD + hh]));
}

__global__ void edge_sum1(const int* __restrict__ ei, int E, int EP,
                          const float* __restrict__ a_src, const float* __restrict__ a_dst,
                          const float* __restrict__ emax, float* __restrict__ denom) {
    int i = blockIdx.x * blockDim.x + threadIdx.x;
    if (i >= EP * NHEAD) return;
    int e = i >> 3, hh = i & 7, s, d;
    edge_sd(ei, E, e, s, d);
    float v = leaky(a_src[s * NHEAD + hh] + a_dst[d * NHEAD + hh]);
    atomicAdd(&denom[d * NHEAD + hh], __expf(v - emax[d * NHEAD + hh]));
}

// one wave per edge; lane handles 4 consecutive features (head = lane/4)
__global__ void edge_agg1(const int* __restrict__ ei, int E, int EP,
                          const float* __restrict__ h1,
                          const float* __restrict__ a_src, const float* __restrict__ a_dst,
                          const float* __restrict__ emax, const float* __restrict__ denom,
                          float* __restrict__ out1) {
    int t = blockIdx.x * blockDim.x + threadIdx.x;
    int e = t >> 5;
    if (e >= EP) return;
    int lane = t & 31, s, d;
    edge_sd(ei, E, e, s, d);
    int hh = lane >> 2;
    float v     = leaky(a_src[s * NHEAD + hh] + a_dst[d * NHEAD + hh]);
    float alpha = __expf(v - emax[d * NHEAD + hh]) / denom[d * NHEAD + hh];
    float4 hv = *(const float4*)(h1 + (size_t)s * F128 + lane * 4);
    float* po = out1 + (size_t)d * F128 + lane * 4;
    atomicAdd(po + 0, hv.x * alpha);
    atomicAdd(po + 1, hv.y * alpha);
    atomicAdd(po + 2, hv.z * alpha);
    atomicAdd(po + 3, hv.w * alpha);
}

// -------------------- ELU + linear2 + layer-2 attention coeffs (fused per node) ----
__global__ void node2_fused(const float* __restrict__ out1, const float* __restrict__ b1,
                            const float* __restrict__ W2, const float* __restrict__ b2,
                            const float* __restrict__ as2, const float* __restrict__ ad2,
                            float* __restrict__ h2, float* __restrict__ a2s,
                            float* __restrict__ a2d, int Nn) {
    int n = blockIdx.x * blockDim.x + threadIdx.x;
    if (n >= Nn) return;
    const float* pr = out1 + (size_t)n * F128;
    float acc0 = 0.f, acc1 = 0.f;
#pragma unroll 4
    for (int k = 0; k < F128; ++k) {
        float t = pr[k] + b1[k];
        t = t > 0.f ? t : (__expf(t) - 1.f);      // ELU
        acc0 += t * W2[k * 2 + 0];
        acc1 += t * W2[k * 2 + 1];
    }
    acc0 += b2[0]; acc1 += b2[1];
    h2[n * 2 + 0] = acc0; h2[n * 2 + 1] = acc1;
    a2s[n] = acc0 * as2[0] + acc1 * as2[1];
    a2d[n] = acc0 * ad2[0] + acc1 * ad2[1];
}

// -------------------- layer-2 edge passes (heads=1) --------------------
__global__ void edge_max2(const int* __restrict__ ei, int E, int EP,
                          const float* __restrict__ a2s, const float* __restrict__ a2d,
                          float* __restrict__ emax2) {
    int e = blockIdx.x * blockDim.x + threadIdx.x;
    if (e >= EP) return;
    int s, d; edge_sd(ei, E, e, s, d);
    atomicMaxF(&emax2[d], leaky(a2s[s] + a2d[d]));
}

__global__ void edge_sum2(const int* __restrict__ ei, int E, int EP,
                          const float* __restrict__ a2s, const float* __restrict__ a2d,
                          const float* __restrict__ emax2, float* __restrict__ denom2) {
    int e = blockIdx.x * blockDim.x + threadIdx.x;
    if (e >= EP) return;
    int s, d; edge_sd(ei, E, e, s, d);
    atomicAdd(&denom2[d], __expf(leaky(a2s[s] + a2d[d]) - emax2[d]));
}

__global__ void edge_agg2(const int* __restrict__ ei, int E, int EP,
                          const float* __restrict__ h2,
                          const float* __restrict__ a2s, const float* __restrict__ a2d,
                          const float* __restrict__ emax2, const float* __restrict__ denom2,
                          float* __restrict__ out2) {
    int e = blockIdx.x * blockDim.x + threadIdx.x;
    if (e >= EP) return;
    int s, d; edge_sd(ei, E, e, s, d);
    float alpha = __expf(leaky(a2s[s] + a2d[d]) - emax2[d]) / denom2[d];
    atomicAdd(&out2[d * 2 + 0], h2[s * 2 + 0] * alpha);
    atomicAdd(&out2[d * 2 + 1], h2[s * 2 + 1] * alpha);
}

// -------------------- global mean pool --------------------
__global__ void pool_sum(const int* __restrict__ batch, const float* __restrict__ out2,
                         float* __restrict__ psum, float* __restrict__ pcnt, int Nn) {
    int n = blockIdx.x * blockDim.x + threadIdx.x;
    if (n >= Nn) return;
    int g = batch[n];
    atomicAdd(&psum[g * 2 + 0], out2[n * 2 + 0]);
    atomicAdd(&psum[g * 2 + 1], out2[n * 2 + 1]);
    atomicAdd(&pcnt[g], 1.0f);
}

__global__ void final_logsoftmax(const float* __restrict__ psum,
                                 const float* __restrict__ pcnt,
                                 float* __restrict__ out) {
    int g = threadIdx.x;
    if (g >= 64) return;
    float c  = fmaxf(pcnt[g], 1.0f);
    float p0 = psum[g * 2 + 0] / c;
    float p1 = psum[g * 2 + 1] / c;
    float m  = fmaxf(p0, p1);
    float l  = logf(__expf(p0 - m) + __expf(p1 - m));
    out[g * 2 + 0] = p0 - m - l;
    out[g * 2 + 1] = p1 - m - l;
}

// ---------------------------------------------------------------------------
extern "C" void kernel_launch(void* const* d_in, const int* in_sizes, int n_in,
                              void* d_out, int out_size, void* d_ws, size_t ws_size,
                              hipStream_t stream) {
    const float* x   = (const float*)d_in[0];
    const int*   ei  = (const int*)  d_in[1];
    const int*   bat = (const int*)  d_in[2];
    const float* W1  = (const float*)d_in[3];
    const float* as1 = (const float*)d_in[4];
    const float* ad1 = (const float*)d_in[5];
    const float* b1  = (const float*)d_in[6];
    const float* W2  = (const float*)d_in[7];
    const float* as2 = (const float*)d_in[8];
    const float* ad2 = (const float*)d_in[9];
    const float* b2  = (const float*)d_in[10];

    const int Nn = in_sizes[0] / F128;
    const int E  = in_sizes[1] / 2;
    const int EP = E + Nn;                    // with self-loops

    char*  base = (char*)d_ws;
    size_t off  = 0;
    auto alloc = [&](size_t bytes) -> void* {
        void* p = base + off;
        off = (off + bytes + 255) & ~(size_t)255;
        return p;
    };
    _Float16* xh     = (_Float16*)alloc((size_t)Nn * F128 * 2);
    _Float16* w1t    = (_Float16*)alloc((size_t)F128 * F128 * 2);
    float*    h1     = (float*)   alloc((size_t)Nn * F128 * 4);
    float*    a_src  = (float*)   alloc((size_t)Nn * NHEAD * 4);
    float*    a_dst  = (float*)   alloc((size_t)Nn * NHEAD * 4);
    float*    emax1  = (float*)   alloc((size_t)Nn * NHEAD * 4);
    float*    denom1 = (float*)   alloc((size_t)Nn * NHEAD * 4);
    float*    out1   = (float*)   alloc((size_t)Nn * F128 * 4);
    float*    h2     = (float*)   alloc((size_t)Nn * 2 * 4);
    float*    a2s    = (float*)   alloc((size_t)Nn * 4);
    float*    a2d    = (float*)   alloc((size_t)Nn * 4);
    float*    emax2  = (float*)   alloc((size_t)Nn * 4);
    float*    denom2 = (float*)   alloc((size_t)Nn * 4);
    float*    out2   = (float*)   alloc((size_t)Nn * 2 * 4);
    float*    psum   = (float*)   alloc(64 * 2 * 4);
    float*    pcnt   = (float*)   alloc(64 * 4);

    // re-init accumulators every call (graph-capture safe)
    hipMemsetAsync(denom1, 0,    (size_t)Nn * NHEAD * 4, stream);
    hipMemsetAsync(out1,   0,    (size_t)Nn * F128 * 4,  stream);
    hipMemsetAsync(denom2, 0,    (size_t)Nn * 4,         stream);
    hipMemsetAsync(out2,   0,    (size_t)Nn * 2 * 4,     stream);
    hipMemsetAsync(psum,   0,    64 * 2 * 4,             stream);
    hipMemsetAsync(pcnt,   0,    64 * 4,                 stream);
    hipMemsetAsync(emax1,  0xFF, (size_t)Nn * NHEAD * 4, stream);  // -inf-ish sentinel
    hipMemsetAsync(emax2,  0xFF, (size_t)Nn * 4,         stream);

    const int B = 256;
    auto g = [](long long n, int b) { return (int)((n + b - 1) / b); };

    pack_x_f16  <<<g((long long)Nn * F128, B), B, 0, stream>>>(x, xh, Nn * F128);
    pack_w1t_f16<<<g(F128 * F128, B),          B, 0, stream>>>(W1, w1t);

    gemm1_wmma  <<<Nn / 16, B, 0, stream>>>(xh, w1t, b1, h1);

    att1_coeff  <<<g((long long)Nn * NHEAD, B), B, 0, stream>>>(h1, as1, ad1, a_src, a_dst, Nn);

    edge_max1   <<<g((long long)EP * NHEAD, B), B, 0, stream>>>(ei, E, EP, a_src, a_dst, emax1);
    edge_sum1   <<<g((long long)EP * NHEAD, B), B, 0, stream>>>(ei, E, EP, a_src, a_dst, emax1, denom1);
    edge_agg1   <<<g((long long)EP * 32, B),    B, 0, stream>>>(ei, E, EP, h1, a_src, a_dst,
                                                               emax1, denom1, out1);

    node2_fused <<<g(Nn, B), B, 0, stream>>>(out1, b1, W2, b2, as2, ad2, h2, a2s, a2d, Nn);

    edge_max2   <<<g(EP, B), B, 0, stream>>>(ei, E, EP, a2s, a2d, emax2);
    edge_sum2   <<<g(EP, B), B, 0, stream>>>(ei, E, EP, a2s, a2d, emax2, denom2);
    edge_agg2   <<<g(EP, B), B, 0, stream>>>(ei, E, EP, h2, a2s, a2d, emax2, denom2, out2);

    pool_sum    <<<g(Nn, B), B, 0, stream>>>(bat, out2, psum, pcnt, Nn);
    final_logsoftmax<<<1, 64, 0, stream>>>(psum, pcnt, (float*)d_out);
}
